// GNNTopK_807453851809
// MI455X (gfx1250) — matrix operand order repaired and checked by reading, not compiled, not run on hardware
//
#include <hip/hip_runtime.h>
#include <hip/hip_bf16.h>

// ---------------- problem constants (match reference) ----------------
#define NNODES   65536      // G * NPER
#define NEDGES   1048576
#define FDIM     128        // in_feats == hidden == 128
#define GGR      64
#define NPER     1024
#define KEEP     512
#define GK       32768      // GGR * KEEP
#define NCLS     10

// ---------------- workspace byte offsets ----------------
#define OFF_XW1    ((size_t)0)            // 32 MB f32 (later: h2bf @0, xw2 @16MB)
#define OFF_H2BF   ((size_t)0)            // 8 MB bf16
#define OFF_XW2    ((size_t)16777216)     // 16 MB f32
#define OFF_H      ((size_t)33554432)     // agg1 -> h  (32 MB f32)
#define OFF_AGG2   ((size_t)67108864)     // xbf (16 MB bf16) then agg2 -> h3 (16 MB f32)
#define OFF_DEG1   ((size_t)83886080)     // N f32
#define OFF_SCORE  ((size_t)84148224)     // N f32
#define OFF_NEWID  ((size_t)84410368)     // N i32
#define OFF_DEG2   ((size_t)84672512)     // GK f32
#define OFF_PERM   ((size_t)84803584)     // GK i32
#define OFF_VALS   ((size_t)84934656)     // GK f32
#define OFF_POOL   ((size_t)85065728)     // G*H f32
#define OFF_PNORM  ((size_t)85098496)     // 1 f32
#define OFF_W1T    ((size_t)85098624)     // 32 KB bf16 (transposed)
#define OFF_W2T    ((size_t)85131392)     // 32 KB bf16 (transposed)

typedef __attribute__((ext_vector_type(16))) __bf16        v16bf;
typedef __attribute__((ext_vector_type(4)))  __bf16        v4bf;
typedef __attribute__((ext_vector_type(8)))  float         v8f;
typedef __attribute__((ext_vector_type(4)))  unsigned int  u32x4;

// load 16 bf16 (two 16-byte chunks) into a WMMA fragment register pair
static __device__ inline v16bf ld_frag(const __bf16* p0, const __bf16* p1) {
  union { u32x4 q[2]; v16bf v; } u;
  u.q[0] = *(const u32x4*)p0;
  u.q[1] = *(const u32x4*)p1;
  return u.v;
}

// ---------------- prep: f32 -> bf16 copy ----------------
__global__ void cvt_bf16_kernel(const float* __restrict__ in, __bf16* __restrict__ out,
                                int n4) {
  int i = blockIdx.x * blockDim.x + threadIdx.x;
  if (i >= n4) return;
  float4 v = ((const float4*)in)[i];
  v4bf o = { (__bf16)v.x, (__bf16)v.y, (__bf16)v.z, (__bf16)v.w };
  ((v4bf*)out)[i] = o;
}

// ---------------- prep: W[k][n] -> Wt[n][k] in bf16 ----------------
__global__ void transpose_w_kernel(const float* __restrict__ W, __bf16* __restrict__ Wt) {
  int i = blockIdx.x * blockDim.x + threadIdx.x;   // 128*128 threads
  int n = i >> 7, k = i & 127;
  Wt[n * FDIM + k] = (__bf16)W[k * FDIM + n];
}

// ---------------- WMMA GEMM: C[M,128] = A[M,128] * B[128,128] --------
// A bf16 row-major, Bt bf16 [N][K] (transposed). One wave computes a
// 32x16 output block (two 16x16 tiles sharing the B fragment); K-loop
// fully unrolled -> 8 v_wmma per wave.
__global__ __launch_bounds__(256) void gemm128_wmma(const __bf16* __restrict__ A,
                                                    const __bf16* __restrict__ Bt,
                                                    float* __restrict__ C,
                                                    int M) {
  const int lane = threadIdx.x & 31;
  const int wave = blockIdx.x * (blockDim.x >> 5) + (threadIdx.x >> 5);
  const int mt = wave >> 3;               // 8 column tiles of 16
  const int nt = wave & 7;
  const int m0 = mt * 32, n0 = nt * 16;
  if (m0 >= M) return;
  const int row = lane & 15;              // A row / B column within tile
  const int hi  = lane >> 4;              // lane half selects K sub-range

  const __bf16* __restrict__ arow0 = A + (size_t)(m0 + row) * FDIM;
  const __bf16* __restrict__ arow1 = arow0 + 16 * FDIM;
  const __bf16* __restrict__ brow  = Bt + (size_t)(n0 + row) * FDIM;

  v8f c0 = {}, c1 = {};
#pragma unroll
  for (int k = 0; k < FDIM; k += 32) {
    const int ka0 = k + hi * 8;           // K 0..7 / 8..15   -> frag[0..7]
    const int ka1 = k + 16 + hi * 8;      // K 16..23 / 24..31 -> frag[8..15]
    v16bf b  = ld_frag(brow + ka0,  brow + ka1);
    v16bf a0 = ld_frag(arow0 + ka0, arow0 + ka1);
    v16bf a1 = ld_frag(arow1 + ka0, arow1 + ka1);
    c0 = __builtin_amdgcn_wmma_f32_16x16x32_bf16(false, a0, false, b,
                                                 (short)0, c0, false, false);
    c1 = __builtin_amdgcn_wmma_f32_16x16x32_bf16(false, a1, false, b,
                                                 (short)0, c1, false, false);
  }
#pragma unroll
  for (int v = 0; v < 8; ++v) {
    C[(size_t)(m0 + v + hi * 8) * FDIM + n0 + row]      = c0[v];
    C[(size_t)(m0 + 16 + v + hi * 8) * FDIM + n0 + row] = c1[v];
  }
}

// ---------------- degree kernels ----------------
__global__ void deg1_kernel(const int* __restrict__ dst, float* __restrict__ deg) {
  int e = blockIdx.x * blockDim.x + threadIdx.x;
  if (e < NEDGES) atomicAdd(&deg[dst[e]], 1.0f);
}
__global__ void deg2_kernel(const int* __restrict__ src, const int* __restrict__ dst,
                            const int* __restrict__ nid, float* __restrict__ deg) {
  int e = blockIdx.x * blockDim.x + threadIdx.x;
  if (e >= NEDGES) return;
  int s = nid[src[e]], d = nid[dst[e]];
  if (s >= 0 && d >= 0) atomicAdd(&deg[d], 1.0f);
}
__global__ void dinv_kernel(float* __restrict__ deg, int n) {
  int i = blockIdx.x * blockDim.x + threadIdx.x;
  if (i < n) deg[i] = rsqrtf(deg[i] + 1.0f);   // deg buffer becomes dinv
}

// ---------------- edge aggregation: wave per edge, float4 per lane ----
__global__ __launch_bounds__(256) void agg1_kernel(const int* __restrict__ src,
                                                   const int* __restrict__ dst,
                                                   const float* __restrict__ dinv,
                                                   const float* __restrict__ xw,
                                                   float* __restrict__ agg) {
  int wave = blockIdx.x * (blockDim.x >> 5) + (threadIdx.x >> 5);
  int lane = threadIdx.x & 31;
  if (wave >= NEDGES) return;
  int s = src[wave], d = dst[wave];
  float coef = dinv[s] * dinv[d];
  float4 v = ((const float4*)(xw + (size_t)s * FDIM))[lane];
  float* ar = agg + (size_t)d * FDIM + lane * 4;
  atomicAdd(ar + 0, coef * v.x);
  atomicAdd(ar + 1, coef * v.y);
  atomicAdd(ar + 2, coef * v.z);
  atomicAdd(ar + 3, coef * v.w);
}
__global__ __launch_bounds__(256) void agg2_kernel(const int* __restrict__ src,
                                                   const int* __restrict__ dst,
                                                   const int* __restrict__ nid,
                                                   const float* __restrict__ dinv,
                                                   const float* __restrict__ xw,
                                                   float* __restrict__ agg) {
  int wave = blockIdx.x * (blockDim.x >> 5) + (threadIdx.x >> 5);
  int lane = threadIdx.x & 31;
  if (wave >= NEDGES) return;
  int s = nid[src[wave]], d = nid[dst[wave]];
  if (s < 0 || d < 0) return;
  float coef = dinv[s] * dinv[d];
  float4 v = ((const float4*)(xw + (size_t)s * FDIM))[lane];
  float* ar = agg + (size_t)d * FDIM + lane * 4;
  atomicAdd(ar + 0, coef * v.x);
  atomicAdd(ar + 1, coef * v.y);
  atomicAdd(ar + 2, coef * v.z);
  atomicAdd(ar + 3, coef * v.w);
}

// ---------------- h = relu(agg + dinv^2 * xw + b)  (in place on agg) --
__global__ void finalize_kernel(float* __restrict__ agg, const float* __restrict__ xw,
                                const float* __restrict__ dinv,
                                const float* __restrict__ bias, int nrows) {
  size_t i = (size_t)blockIdx.x * blockDim.x + threadIdx.x;
  if (i >= (size_t)nrows * FDIM) return;
  int r = (int)(i >> 7), f = (int)(i & 127);
  float di = dinv[r];
  float v = agg[i] + di * di * xw[i] + bias[f];
  agg[i] = v > 0.f ? v : 0.f;
}

// ---------------- score = tanh(h.p / ||p||) ----------------
__global__ void pnorm_kernel(const float* __restrict__ p, float* __restrict__ out) {
  int lane = threadIdx.x;
  float s = 0.f;
  for (int i = lane; i < FDIM; i += 32) s += p[i] * p[i];
  for (int off = 16; off; off >>= 1) s += __shfl_xor(s, off, 32);
  if (lane == 0) out[0] = rsqrtf(s);            // 1/||p||
}
__global__ __launch_bounds__(256) void score_kernel(const float* __restrict__ h,
                                                    const float* __restrict__ p,
                                                    const float* __restrict__ pinv,
                                                    float* __restrict__ score) {
  int wave = blockIdx.x * (blockDim.x >> 5) + (threadIdx.x >> 5);
  int lane = threadIdx.x & 31;
  if (wave >= NNODES) return;
  float4 a = ((const float4*)(h + (size_t)wave * FDIM))[lane];
  float4 b = ((const float4*)p)[lane];
  float s = a.x * b.x + a.y * b.y + a.z * b.z + a.w * b.w;
  for (int off = 16; off; off >>= 1) s += __shfl_xor(s, off, 32);
  if (lane == 0) score[wave] = tanhf(s * pinv[0]);
}

// ---------------- per-graph top-512-of-1024 via LDS bitonic sort ------
__global__ __launch_bounds__(1024) void topk_kernel(const float* __restrict__ score,
                                                    int* __restrict__ newid,
                                                    int* __restrict__ perm,
                                                    float* __restrict__ vals) {
  __shared__ float sv[NPER];
  __shared__ int   si[NPER];
  int g = blockIdx.x, t = threadIdx.x;
  sv[t] = score[g * NPER + t];
  si[t] = t;
  __syncthreads();
  for (int ksz = 2; ksz <= NPER; ksz <<= 1) {
    for (int j = ksz >> 1; j > 0; j >>= 1) {
      int partner = t ^ j;
      if (partner > t) {
        float v0 = sv[t], v1 = sv[partner];
        int   i0 = si[t], i1 = si[partner];
        // descending by value, ascending index tie-break (strict order)
        bool after = (v0 < v1) || (v0 == v1 && i0 > i1);
        bool dirDesc = ((t & ksz) == 0);
        bool doSwap = dirDesc ? after : !after;
        if (doSwap) { sv[t] = v1; sv[partner] = v0; si[t] = i1; si[partner] = i0; }
      }
      __syncthreads();
    }
  }
  if (t < KEEP) {
    int node = g * NPER + si[t];
    int nid  = g * KEEP + t;
    newid[node] = nid;
    perm[nid]   = node;
    vals[nid]   = sv[t];
  }
}

// ---------------- h2(bf16) = h[perm] * vals ----------------
__global__ __launch_bounds__(256) void gather_kernel(const float* __restrict__ h,
                                                     const int* __restrict__ perm,
                                                     const float* __restrict__ vals,
                                                     __bf16* __restrict__ h2) {
  int wave = blockIdx.x * (blockDim.x >> 5) + (threadIdx.x >> 5);
  int lane = threadIdx.x & 31;
  if (wave >= GK) return;
  int node = perm[wave];
  float v = vals[wave];
  float4 a = ((const float4*)(h + (size_t)node * FDIM))[lane];
  v4bf o = { (__bf16)(a.x * v), (__bf16)(a.y * v),
             (__bf16)(a.z * v), (__bf16)(a.w * v) };
  ((v4bf*)(h2 + (size_t)wave * FDIM))[lane] = o;
}

// ---------------- global mean pool over KEEP rows per graph ----------
__global__ void meanpool_kernel(const float* __restrict__ h3, float* __restrict__ pooled) {
  int g = blockIdx.x, f = threadIdx.x;
  const float* base = h3 + (size_t)g * KEEP * FDIM + f;
  float s = 0.f;
  for (int j = 0; j < KEEP; ++j) s += base[(size_t)j * FDIM];
  pooled[g * FDIM + f] = s * (1.0f / KEEP);
}

// ---------------- out = pooled @ Wl + bl ----------------
__global__ void classifier_kernel(const float* __restrict__ pooled,
                                  const float* __restrict__ Wl,
                                  const float* __restrict__ bl,
                                  float* __restrict__ out) {
  int i = blockIdx.x * blockDim.x + threadIdx.x;
  if (i >= GGR * NCLS) return;
  int g = i / NCLS, c = i % NCLS;
  float s = bl[c];
  for (int f = 0; f < FDIM; ++f) s += pooled[g * FDIM + f] * Wl[f * NCLS + c];
  out[i] = s;
}

// =====================================================================
extern "C" void kernel_launch(void* const* d_in, const int* in_sizes, int n_in,
                              void* d_out, int out_size, void* d_ws, size_t ws_size,
                              hipStream_t stream) {
  (void)in_sizes; (void)n_in; (void)out_size; (void)ws_size;
  const float* x   = (const float*)d_in[0];
  const int*   ei  = (const int*)d_in[1];          // [2, E]
  const float* W1  = (const float*)d_in[3];
  const float* b1  = (const float*)d_in[4];
  const float* p   = (const float*)d_in[5];
  const float* W2  = (const float*)d_in[6];
  const float* b2  = (const float*)d_in[7];
  const float* Wl  = (const float*)d_in[8];
  const float* bl  = (const float*)d_in[9];
  const int* src = ei;
  const int* dst = ei + NEDGES;

  char* ws = (char*)d_ws;
  float*  xw1   = (float*) (ws + OFF_XW1);
  __bf16* h2bf  = (__bf16*)(ws + OFF_H2BF);
  float*  xw2   = (float*) (ws + OFF_XW2);
  float*  h     = (float*) (ws + OFF_H);      // agg1 then h (in place)
  __bf16* xbf   = (__bf16*)(ws + OFF_AGG2);   // bf16 x, dead after gemm1
  float*  h3    = (float*) (ws + OFF_AGG2);   // then agg2 -> h3 (in place)
  float*  dinv1 = (float*) (ws + OFF_DEG1);
  float*  score = (float*) (ws + OFF_SCORE);
  int*    newid = (int*)   (ws + OFF_NEWID);
  float*  dinv2 = (float*) (ws + OFF_DEG2);
  int*    perm  = (int*)   (ws + OFF_PERM);
  float*  vals  = (float*) (ws + OFF_VALS);
  float*  pool  = (float*) (ws + OFF_POOL);
  float*  pinv  = (float*) (ws + OFF_PNORM);
  __bf16* W1t   = (__bf16*)(ws + OFF_W1T);
  __bf16* W2t   = (__bf16*)(ws + OFF_W2T);
  float*  out   = (float*)d_out;

  // ---- init accumulators (graph-capture-safe memsets) ----
  hipMemsetAsync(h,     0,    (size_t)NNODES * FDIM * 4, stream);  // agg1
  hipMemsetAsync(dinv1, 0,    (size_t)NNODES * 4,        stream);  // deg1
  hipMemsetAsync(dinv2, 0,    (size_t)GK * 4,            stream);  // deg2
  hipMemsetAsync(newid, 0xFF, (size_t)NNODES * 4,        stream);  // -1

  // ---- prep: bf16 operands for WMMA ----
  cvt_bf16_kernel<<<(NNODES * FDIM / 4) / 256, 256, 0, stream>>>(x, xbf, NNODES * FDIM / 4);
  transpose_w_kernel<<<FDIM * FDIM / 256, 256, 0, stream>>>(W1, W1t);
  transpose_w_kernel<<<FDIM * FDIM / 256, 256, 0, stream>>>(W2, W2t);

  // ---- layer 1: xw1 = x @ W1 (WMMA, 32x16 per wave) ----
  gemm128_wmma<<<(NNODES / 32) * 8 / 8, 256, 0, stream>>>(xbf, W1t, xw1, NNODES);
  deg1_kernel<<<NEDGES / 256, 256, 0, stream>>>(dst, dinv1);
  dinv_kernel<<<NNODES / 256, 256, 0, stream>>>(dinv1, NNODES);
  agg1_kernel<<<NEDGES / 8, 256, 0, stream>>>(src, dst, dinv1, xw1, h);
  finalize_kernel<<<(NNODES * FDIM) / 256, 256, 0, stream>>>(h, xw1, dinv1, b1, NNODES);

  // ---- TopK pooling ----
  pnorm_kernel<<<1, 32, 0, stream>>>(p, pinv);
  score_kernel<<<NNODES / 8, 256, 0, stream>>>(h, p, pinv, score);
  topk_kernel<<<GGR, NPER, 0, stream>>>(score, newid, perm, vals);
  gather_kernel<<<GK / 8, 256, 0, stream>>>(h, perm, vals, h2bf);

  // ---- layer 2: xw2 = h2 @ W2 (WMMA) ----
  gemm128_wmma<<<(GK / 32) * 8 / 8, 256, 0, stream>>>(h2bf, W2t, xw2, GK);
  hipMemsetAsync(h3, 0, (size_t)GK * FDIM * 4, stream);            // agg2 (xbf now dead)
  deg2_kernel<<<NEDGES / 256, 256, 0, stream>>>(src, dst, newid, dinv2);
  dinv_kernel<<<GK / 256, 256, 0, stream>>>(dinv2, GK);
  agg2_kernel<<<NEDGES / 8, 256, 0, stream>>>(src, dst, newid, dinv2, xw2, h3);
  finalize_kernel<<<(GK * FDIM) / 256, 256, 0, stream>>>(h3, xw2, dinv2, b2, GK);

  // ---- readout ----
  meanpool_kernel<<<GGR, FDIM, 0, stream>>>(h3, pool);
  classifier_kernel<<<(GGR * NCLS + 255) / 256, 256, 0, stream>>>(pool, Wl, bl, out);
}
